// DynamicGraphAttention_60301340836408
// MI455X (gfx1250) — matrix operand order repaired and checked by reading, not compiled, not run on hardware
//
#include <hip/hip_runtime.h>
#include <math.h>

// ---------------------------------------------------------------------------
// Problem constants (match reference)
// ---------------------------------------------------------------------------
#define D_MODEL 256
#define N_NODES 2048
#define BATCH   16
// ETA = 1.0, EPS = 1e-3, scale = 1/sqrt(256) = 1/16

typedef __bf16 bf16_t;
typedef __attribute__((ext_vector_type(16))) __bf16 v16bf;
typedef __attribute__((ext_vector_type(8)))  float  v8f;
typedef unsigned int v4u __attribute__((ext_vector_type(4)));
typedef int          v8i __attribute__((ext_vector_type(8)));
typedef int          v4i __attribute__((ext_vector_type(4)));

// ---------------------------------------------------------------------------
// WMMA fragment loaders (CDNA5 wave32 layouts, CDNA5 ISA §7.12.2)
//
// A 16x32 bf16 (MxK): lane l -> row m = l&15, half kh = l>>4.
//   frag elements 0..7  : K = 8*kh + e
//   frag elements 8..15 : K = 8*kh + 16 + (e-8)
// B 32x16 bf16 (KxN): lane l -> col n = l&15, half kh = l>>4.
//   frag elements 0..7  : K = 8*kh + e; 8..15 : K = 8*kh + 16 + (e-8)
// C/D 16x16 f32: lane l -> col n = l&15; VGPR r -> row m = r + 8*(l>>4)
// ---------------------------------------------------------------------------

__device__ __forceinline__ v16bf load_a_frag_any(const bf16_t* src, int ld,
                                                 int m, int kh, int dk) {
    const bf16_t* p = src + (size_t)m * ld + dk + kh * 8;
    v16bf a;
#pragma unroll
    for (int e = 0; e < 8; ++e) {
        a[e]     = p[e];        // K = 8*kh + e
        a[e + 8] = p[16 + e];   // K = 8*kh + 16 + e
    }
    return a;
}

__device__ __forceinline__ v16bf load_a_frag_f32(const float* __restrict__ src,
                                                 int ld, int m, int kh, int dk) {
    const float* p = src + (size_t)m * ld + dk + kh * 8;
    v16bf a;
#pragma unroll
    for (int e = 0; e < 8; ++e) {
        a[e]     = (bf16_t)p[e];
        a[e + 8] = (bf16_t)p[16 + e];
    }
    return a;
}

// B operand: column n of the WMMA B matrix is a contiguous row of `src`
// (works for both W^T in the projection and K^T in Q@K^T).
__device__ __forceinline__ v16bf load_b_frag_bf16(const bf16_t* __restrict__ src,
                                                  int ld, int n, int kh, int dk) {
    const bf16_t* p = src + (size_t)n * ld + dk + kh * 8;
    v16bf b;
#pragma unroll
    for (int e = 0; e < 8; ++e) { b[e] = p[e]; b[e + 8] = p[16 + e]; }
    return b;
}

__device__ __forceinline__ v8f wmma_bf16(v16bf a, v16bf b, v8f c) {
    return __builtin_amdgcn_wmma_f32_16x16x32_bf16(
        /*neg_a=*/false, a, /*neg_b=*/false, b,
        /*c_mod=*/(short)0, c, /*reuse_a=*/false, /*reuse_b=*/false);
}

// ---------------------------------------------------------------------------
// Kernel 0a: fused bias = logit(clip(A_stat)) with mask==0 -> -inf  [N x N] f32
// ---------------------------------------------------------------------------
__global__ __launch_bounds__(256) void build_bias_kernel(
    const float* __restrict__ A, const int* __restrict__ M,
    float* __restrict__ Bias, int n) {
    int i = blockIdx.x * blockDim.x + threadIdx.x;
    int stride = gridDim.x * blockDim.x;
    for (; i < n; i += stride) {
        float a = fminf(fmaxf(A[i], 1e-3f), 1.0f - 1e-3f);
        float bias = logf(a) - log1pf(-a);    // ETA = 1.0
        if (M[i] == 0) bias = -INFINITY;
        Bias[i] = bias;
    }
}

// ---------------------------------------------------------------------------
// Kernel 0b: Wq/Wk f32 -> bf16 (one-time, keeps the projection hot loop
// free of v_cvt chains and halves its operand bytes)
// ---------------------------------------------------------------------------
__global__ __launch_bounds__(256) void convert_w_kernel(
    const float* __restrict__ Wq, const float* __restrict__ Wk,
    bf16_t* __restrict__ Wqb, bf16_t* __restrict__ Wkb, int n) {
    int i = blockIdx.x * blockDim.x + threadIdx.x;
    if (i < n) {
        Wqb[i] = (bf16_t)Wq[i];
        Wkb[i] = (bf16_t)Wk[i];
    }
}

// ---------------------------------------------------------------------------
// Kernel 1: Q = H @ Wq^T, K = H @ Wk^T  -> bf16 workspace
// One wave: 16 rows x 64 cols of BOTH Q and K. All 8 B fragments of a
// k-step are loaded into distinct registers up front so the 16 b128 loads
// issue back-to-back and are covered by graduated s_wait_loadcnt.
// ---------------------------------------------------------------------------
__global__ __launch_bounds__(256) void proj_qk_kernel(
    const float* __restrict__ Hn, const bf16_t* __restrict__ Wqb,
    const bf16_t* __restrict__ Wkb, bf16_t* __restrict__ Qb,
    bf16_t* __restrict__ Kb) {
    const int wid  = (blockIdx.x * blockDim.x + threadIdx.x) >> 5;
    const int lane = threadIdx.x & 31;
    const int mt = wid >> 2;          // 16-row tile index over B*N rows
    const int eg = wid & 3;           // 64-col group of output features
    const int e0 = eg * 64;
    const int mn = lane & 15;
    const int kh = lane >> 4;

    const float* Hrow = Hn + (size_t)mt * 16 * D_MODEL;

    v8f accQ[4], accK[4];
#pragma unroll
    for (int t = 0; t < 4; ++t) { accQ[t] = (v8f){}; accK[t] = (v8f){}; }

    for (int dk = 0; dk < D_MODEL; dk += 32) {
        v16bf a = load_a_frag_f32(Hrow, D_MODEL, mn, kh, dk);
        v16bf bq[4], bk[4];
#pragma unroll
        for (int t = 0; t < 4; ++t) {
            bq[t] = load_b_frag_bf16(Wqb, D_MODEL, e0 + t * 16 + mn, kh, dk);
            bk[t] = load_b_frag_bf16(Wkb, D_MODEL, e0 + t * 16 + mn, kh, dk);
        }
#pragma unroll
        for (int t = 0; t < 4; ++t) {
            accQ[t] = wmma_bf16(a, bq[t], accQ[t]);
            accK[t] = wmma_bf16(a, bk[t], accK[t]);
        }
    }

    bf16_t* qdst = Qb + (size_t)mt * 16 * D_MODEL;
    bf16_t* kdst = Kb + (size_t)mt * 16 * D_MODEL;
#pragma unroll
    for (int t = 0; t < 4; ++t) {
#pragma unroll
        for (int r = 0; r < 8; ++r) {
            int row = r + 8 * kh;                 // C/D layout
            int col = e0 + t * 16 + mn;
            qdst[(size_t)row * D_MODEL + col] = (bf16_t)accQ[t][r];
            kdst[(size_t)row * D_MODEL + col] = (bf16_t)accK[t][r];
        }
    }
}

// ---------------------------------------------------------------------------
// Kernel 2: logits = (Q K^T)/16 + Bias, row softmax, write [B,N,N] f32.
// One block (512 thr = 16 waves) per (batch, 16-query tile).
//  - The contiguous 8KB Q tile is DMA'd to LDS by the Tensor Data Mover
//    (tensor_load_to_lds, wave 0) and A-fragments come from ds_load.
//  - Wave w owns k-columns [w*128, w*128+128): 8 register accumulators,
//    all 8 K fragments of a k-step loaded up front (16 b128 loads in flight).
//  - Row max / exp-sum: shfl_xor butterfly + 2KB LDS split reduction.
//  - Output written with non-temporal stores (write-once, 256MB) so the
//    L2-resident Q/K/Bias working set is not evicted.
// ---------------------------------------------------------------------------
__global__ __launch_bounds__(512) void attn_softmax_kernel(
    const bf16_t* __restrict__ Qb, const bf16_t* __restrict__ Kb,
    const float* __restrict__ Bias, float* __restrict__ out) {
    // single shared buffer => starts at LDS offset 0 (used by the TDM D#)
    __shared__ __align__(128) unsigned char lds_raw[8192 + 2 * 16 * 16 * 4];
    bf16_t* shQ  = (bf16_t*)lds_raw;                   // 16 x 256 bf16, 8KB
    float*  smax = (float*)(lds_raw + 8192);           // [16 rows][16 waves]
    float*  ssum = (float*)(lds_raw + 8192 + 1024);    // [16 rows][16 waves]

    const int b    = blockIdx.x >> 7;       // N/16 = 128 tiles per batch
    const int qt   = blockIdx.x & 127;
    const int q0   = qt * 16;
    const int w    = threadIdx.x >> 5;      // wave 0..15
    const int lane = threadIdx.x & 31;
    const int mn   = lane & 15;
    const int kh   = lane >> 4;
    const int k0w  = w * 128;               // 8 k-tiles per wave

    const bf16_t* Qrow = Qb + ((size_t)b * N_NODES + q0) * D_MODEL;
    const bf16_t* Krow = Kb + (size_t)b * N_NODES * D_MODEL;

    // --- TDM: wave 0 DMAs the 16x256 bf16 Q tile into LDS ------------------
    if (threadIdx.x < 32) {
        unsigned long long ga = (unsigned long long)(uintptr_t)Qrow;
        v4u g0;
        g0[0] = 1u;                                   // count=1 (valid user D#)
        g0[1] = 0u;                                   // lds_addr = 0 (shQ base)
        g0[2] = (unsigned)(ga & 0xFFFFFFFFu);         // global_addr[31:0]
        g0[3] = (unsigned)((ga >> 32) & 0x01FFFFFFu)  // global_addr[56:32]
                | 0x80000000u;                        // type=2 ("image")
        v8i g1;
        g1[0] = (int)(1u << 16);      // workgroup_mask=0, data_size=1 (2B)
        g1[1] = (int)(256u << 16);    // tensor_dim0 = 256 (low 16)
        g1[2] = (int)(16u << 16);     // tensor_dim0 hi=0 | tensor_dim1 = 16
        g1[3] = (int)(256u << 16);    // tensor_dim1 hi=0 | tile_dim0 = 256
        g1[4] = 16;                   // tile_dim1 = 16, tile_dim2 = 0 (2D)
        g1[5] = 256;                  // tensor_dim0_stride = 256 (low 32)
        g1[6] = 0;                    // stride hi / dim1_stride (unused, 2D)
        g1[7] = 0;
        v4i gz4 = {0, 0, 0, 0};       // groups 2/3 unused for 2D tiles
        v8i gz8 = {0, 0, 0, 0, 0, 0, 0, 0};
        __builtin_amdgcn_tensor_load_to_lds(g0, g1, gz4, gz4, gz8, 0);
        __builtin_amdgcn_s_wait_tensorcnt(0);
    }
    __syncthreads();

    // --- logits: 8 WMMA tiles per wave, K fragments batch-loaded ------------
    v8f acc[8];
#pragma unroll
    for (int t = 0; t < 8; ++t) acc[t] = (v8f){};

    for (int dk = 0; dk < D_MODEL; dk += 32) {
        v16bf a = load_a_frag_any(shQ, D_MODEL, mn, kh, dk);   // ds_load
        v16bf bfr[8];
#pragma unroll
        for (int t = 0; t < 8; ++t)
            bfr[t] = load_b_frag_bf16(Krow, D_MODEL, k0w + t * 16 + mn, kh, dk);
#pragma unroll
        for (int t = 0; t < 8; ++t)
            acc[t] = wmma_bf16(a, bfr[t], acc[t]);
    }

    // --- scale + bias (mask folded as -inf), per-row max --------------------
    const float scale = 0.0625f;  // 1/sqrt(256)
    float vmax[8];
#pragma unroll
    for (int r = 0; r < 8; ++r) vmax[r] = -INFINITY;

#pragma unroll
    for (int t = 0; t < 8; ++t) {
#pragma unroll
        for (int r = 0; r < 8; ++r) {
            int row = q0 + r + 8 * kh;
            int col = k0w + t * 16 + mn;
            float v = acc[t][r] * scale + Bias[(size_t)row * N_NODES + col];
            acc[t][r] = v;
            vmax[r] = fmaxf(vmax[r], v);
        }
    }
#pragma unroll
    for (int off = 1; off < 16; off <<= 1) {
#pragma unroll
        for (int r = 0; r < 8; ++r)
            vmax[r] = fmaxf(vmax[r], __shfl_xor(vmax[r], off, 32));
    }
    if (mn == 0) {
#pragma unroll
        for (int r = 0; r < 8; ++r) smax[(r + 8 * kh) * 16 + w] = vmax[r];
    }
    __syncthreads();

    float rmax[8];
#pragma unroll
    for (int r = 0; r < 8; ++r) {
        float m = -INFINITY;
#pragma unroll
        for (int ww = 0; ww < 16; ++ww)
            m = fmaxf(m, smax[(r + 8 * kh) * 16 + ww]);
        rmax[r] = m;
    }

    // --- exp + per-row sums --------------------------------------------------
    float vsum[8];
#pragma unroll
    for (int r = 0; r < 8; ++r) vsum[r] = 0.0f;
#pragma unroll
    for (int t = 0; t < 8; ++t) {
#pragma unroll
        for (int r = 0; r < 8; ++r) {
            float e = __expf(acc[t][r] - rmax[r]);
            acc[t][r] = e;
            vsum[r] += e;
        }
    }
#pragma unroll
    for (int off = 1; off < 16; off <<= 1) {
#pragma unroll
        for (int r = 0; r < 8; ++r)
            vsum[r] += __shfl_xor(vsum[r], off, 32);
    }
    if (mn == 0) {
#pragma unroll
        for (int r = 0; r < 8; ++r) ssum[(r + 8 * kh) * 16 + w] = vsum[r];
    }
    __syncthreads();

    float rinv[8];
#pragma unroll
    for (int r = 0; r < 8; ++r) {
        float s = 0.0f;
#pragma unroll
        for (int ww = 0; ww < 16; ++ww) s += ssum[(r + 8 * kh) * 16 + ww];
        rinv[r] = 1.0f / s;
    }

    // --- normalized write-out: non-temporal (write-once 256MB) --------------
    float* orow = out + ((size_t)b * N_NODES + q0) * N_NODES;
#pragma unroll
    for (int t = 0; t < 8; ++t) {
#pragma unroll
        for (int r = 0; r < 8; ++r) {
            int row = r + 8 * kh;
            int col = k0w + t * 16 + mn;
            __builtin_nontemporal_store(acc[t][r] * rinv[r],
                                        &orow[(size_t)row * N_NODES + col]);
        }
    }
}

// ---------------------------------------------------------------------------
// Launch
// ---------------------------------------------------------------------------
extern "C" void kernel_launch(void* const* d_in, const int* in_sizes, int n_in,
                              void* d_out, int out_size, void* d_ws, size_t ws_size,
                              hipStream_t stream) {
    (void)in_sizes; (void)n_in; (void)out_size; (void)ws_size;

    const float* Hn     = (const float*)d_in[0];   // [B, N, D] f32
    const float* A_stat = (const float*)d_in[1];   // [N, N]   f32
    const int*   M_mask = (const int*)d_in[2];     // [N, N]   i32
    const float* Wq     = (const float*)d_in[3];   // [D, D]   f32
    const float* Wk     = (const float*)d_in[4];   // [D, D]   f32
    float*       out    = (float*)d_out;           // [B, N, N] f32

    // workspace: Q bf16 | K bf16 | Bias f32 | Wq bf16 | Wk bf16  (~51 MB)
    const size_t qk_elems = (size_t)BATCH * N_NODES * D_MODEL;
    const size_t w_elems  = (size_t)D_MODEL * D_MODEL;
    bf16_t* Qb   = (bf16_t*)d_ws;
    bf16_t* Kb   = Qb + qk_elems;
    float*  Bias = (float*)(Kb + qk_elems);
    bf16_t* Wqb  = (bf16_t*)(Bias + (size_t)N_NODES * N_NODES);
    bf16_t* Wkb  = Wqb + w_elems;

    // 0a) bias/mask fusion: 4M elements
    build_bias_kernel<<<4096, 256, 0, stream>>>(A_stat, M_mask, Bias,
                                                N_NODES * N_NODES);
    // 0b) weights to bf16: 64K elements
    convert_w_kernel<<<(int)(w_elems / 256), 256, 0, stream>>>(
        Wq, Wk, Wqb, Wkb, (int)w_elems);

    // 1) projections: 2048 m-tiles * 4 e-groups = 8192 waves / 8 per block
    proj_qk_kernel<<<1024, 256, 0, stream>>>(Hn, Wqb, Wkb, Qb, Kb);

    // 2) attention logits + softmax: one block per (batch, 16-query tile)
    attn_softmax_kernel<<<BATCH * (N_NODES / 16), 512, 0, stream>>>(Qb, Kb, Bias,
                                                                    out);
}
